// MultiHeadSelfAttention_34419867910632
// MI455X (gfx1250) — matrix operand order repaired and checked by reading, not compiled, not run on hardware
//
#include <hip/hip_runtime.h>
#include <hip/hip_bf16.h>

typedef __attribute__((ext_vector_type(16))) _Float16 v16h;
typedef __attribute__((ext_vector_type(8)))  _Float16 v8h;
typedef __attribute__((ext_vector_type(8)))  float    v8f;

#define BATCH  2
#define SEQ    2048
#define DMODEL 1024
#define NH     32
#define HD     32
#define QTILES (SEQ / 16)   // 128
#define KBLK   64           // keys staged per LDS block

// ---------- helpers ----------
__device__ __forceinline__ float rowmax16(float v) {
    #pragma unroll
    for (int i = 1; i < 16; i <<= 1) v = fmaxf(v, __shfl_xor(v, i, 32));
    return v;
}
__device__ __forceinline__ float rowsum16(float v) {
    #pragma unroll
    for (int i = 1; i < 16; i <<= 1) v += __shfl_xor(v, i, 32);
    return v;
}
// Assemble WMMA f16 A-operand (16x32, row M = lane&15) from one contiguous
// 16-half chunk at K = 2*base8 (contiguous k order, matching B).
__device__ __forceinline__ v16h make_a(const _Float16* row, int base8) {
    return *(const v16h*)(row + 2 * base8);
}
// gfx1250 async global->LDS copy (16B per lane), tracked by ASYNCcnt.
// Generic LDS pointers carry the LDS byte offset in their low 32 bits.
__device__ __forceinline__ void async_b128(const _Float16* ldsDst, const _Float16* gSrc) {
    unsigned off = (unsigned)(uintptr_t)ldsDst;
    asm volatile("global_load_async_to_lds_b128 %0, %1, off"
                 :: "v"(off), "v"(gSrc) : "memory");
}
__device__ __forceinline__ void wait_async() {
    asm volatile("s_wait_asynccnt 0x0" ::: "memory");
}

// ---------- weight prep: transpose + fp32->f16 ----------
__global__ void prep_small(const float* __restrict__ wq, const float* __restrict__ wk,
                           const float* __restrict__ wv,
                           _Float16* __restrict__ wqT, _Float16* __restrict__ wkT,
                           _Float16* __restrict__ wvT) {
    int t = threadIdx.x;            // 1024 threads
    int i = t >> 5, o = t & 31;     // wT[out][in] = w[in][out]
    wqT[o * 32 + i] = (_Float16)wq[i * 32 + o];
    wkT[o * 32 + i] = (_Float16)wk[i * 32 + o];
    wvT[o * 32 + i] = (_Float16)wv[i * 32 + o];
}
__global__ void prep_wo(const float* __restrict__ wo, _Float16* __restrict__ woT) {
    size_t g = (size_t)blockIdx.x * 256 + threadIdx.x;   // 1M threads
    size_t k = g >> 10, n = g & 1023;
    woT[n * 1024 + k] = (_Float16)wo[g];                 // woT[out][in]
}

// ---------- QKV projection: per wave, 16 rows of one (b,h) ----------
__global__ __launch_bounds__(32)
void qkv_kernel(const float* __restrict__ x,
                const _Float16* __restrict__ wqT, const float* __restrict__ bq,
                const _Float16* __restrict__ wkT, const float* __restrict__ bk,
                const _Float16* __restrict__ wvT, const float* __restrict__ bv,
                _Float16* __restrict__ Q, _Float16* __restrict__ Kb,
                _Float16* __restrict__ Vt) {
    const int lane = threadIdx.x;
    const int mrow = lane & 15, hi = lane >> 4, base8 = hi * 8;
    const int idx = blockIdx.x;
    const int qt = idx & (QTILES - 1);
    const int h  = (idx >> 7) & (NH - 1);
    const int b  = idx >> 12;
    const int s  = qt * 16 + mrow;
    const float scale = 0.17677669529663689f;   // 1/sqrt(32)

    const float* xp = x + ((size_t)(b * SEQ + s)) * DMODEL + h * HD;
    v16h a;
    #pragma unroll
    for (int e = 0; e < 8; e++) a[e]     = (_Float16)xp[2 * base8 + e];
    #pragma unroll
    for (int e = 0; e < 8; e++) a[e + 8] = (_Float16)xp[2 * base8 + 8 + e];

    const size_t qkbase = ((size_t)(b * NH + h)) * SEQ * HD;
    v8f zero = {};

    #pragma unroll
    for (int t = 0; t < 2; t++) {   // output-dim column tiles 0-15 / 16-31
        const int koff = hi * 16;
        const int n = t * 16 + mrow;
        v16h bwq = *(const v16h*)(wqT + n * HD + koff);
        v16h bwk = *(const v16h*)(wkT + n * HD + koff);
        v16h bwv = *(const v16h*)(wvT + n * HD + koff);
        v8f aq = __builtin_amdgcn_wmma_f32_16x16x32_f16(false, a, false, bwq, (short)0, zero, false, false);
        v8f ak = __builtin_amdgcn_wmma_f32_16x16x32_f16(false, a, false, bwk, (short)0, zero, false, false);
        v8f av = __builtin_amdgcn_wmma_f32_16x16x32_f16(false, a, false, bwv, (short)0, zero, false, false);
        const float biq = bq[n], bik = bk[n], biv = bv[n];
        #pragma unroll
        for (int r = 0; r < 8; r++) {
            const int row = qt * 16 + hi * 8 + r;   // C layout: M = r + hi*8
            const int dim = n;
            Q[qkbase + (size_t)row * HD + dim]  = (_Float16)((aq[r] + biq) * scale);
            Kb[qkbase + (size_t)row * HD + dim] = (_Float16)(ak[r] + bik);
            Vt[qkbase + (size_t)dim * SEQ + row] = (_Float16)(av[r] + biv);  // transposed
        }
    }
}

// ---------- fused flash attention ----------
// 256 threads = 8 waves; block covers 128 queries of one (b,h); K/V staged into
// LDS (double-buffered, async DMA) and shared by all 8 waves.
__global__ __launch_bounds__(256)
void attn_kernel(const _Float16* __restrict__ Q, const _Float16* __restrict__ Kb,
                 const _Float16* __restrict__ Vt, _Float16* __restrict__ attn) {
    __shared__ _Float16 kLds[2][KBLK * HD];     // [key][dim]   4KB x2
    __shared__ _Float16 vLds[2][HD * KBLK];     // [dim][key]   4KB x2
    __shared__ _Float16 pLds[8][16 * KBLK];     // per-wave P   2KB x8

    const int tid  = threadIdx.x;
    const int lane = tid & 31, wave = tid >> 5;
    const int mrow = lane & 15, hi = lane >> 4, base8 = hi * 8;
    const int idx = blockIdx.x;
    const int qt = idx & 15;            // SEQ/128 = 16 blocks per (b,h)
    const int h  = (idx >> 4) & (NH - 1);
    const int b  = idx >> 9;
    const size_t qkbase = ((size_t)(b * NH + h)) * SEQ * HD;
    const _Float16* Kg = Kb + qkbase;
    const _Float16* Vg = Vt + qkbase;

    // per-thread staging slots: K = 2048 halves linear; V = 32 dims x 8 chunks
    const int vdim = tid >> 3, vchk = (tid & 7) * 8;

    const int qrow = qt * 128 + wave * 16;
    const _Float16* qp = Q + qkbase + (size_t)(qrow + mrow) * HD;
    const v16h aq = make_a(qp, base8);

    float m[8], l[8];
    v8f o0 = {}, o1 = {};
    #pragma unroll
    for (int r = 0; r < 8; r++) { m[r] = -1e30f; l[r] = 0.0f; }

    // prologue: stage first K/V block
    async_b128(&kLds[0][tid * 8], Kg + tid * 8);
    async_b128(&vLds[0][vdim * KBLK + vchk], Vg + (size_t)vdim * SEQ + vchk);

    int buf = 0;
    for (int kbeg = 0; kbeg < SEQ; kbeg += KBLK, buf ^= 1) {
        wait_async();
        __syncthreads();                       // staged block visible to all waves
        if (kbeg + KBLK < SEQ) {               // prefetch next block into other buffer
            const int nk = kbeg + KBLK;
            async_b128(&kLds[buf ^ 1][tid * 8], Kg + (size_t)nk * HD + tid * 8);
            async_b128(&vLds[buf ^ 1][vdim * KBLK + vchk],
                       Vg + (size_t)vdim * SEQ + nk + vchk);
        }
        const _Float16* kB = kLds[buf];
        const _Float16* vB = vLds[buf];

        // scores for 16 queries x 64 keys: 4 WMMAs
        v8f s[4];
        #pragma unroll
        for (int j = 0; j < 4; j++) {
            v16h bk = *(const v16h*)(kB + (j * 16 + mrow) * HD + hi * 16);
            v8f z = {};
            s[j] = __builtin_amdgcn_wmma_f32_16x16x32_f16(false, aq, false, bk, (short)0, z, false, false);
        }

        // online softmax; one shuffle-tree per row covers all 64 keys
        _Float16* pW = pLds[wave];
        #pragma unroll
        for (int r = 0; r < 8; r++) {
            float mt = fmaxf(fmaxf(s[0][r], s[1][r]), fmaxf(s[2][r], s[3][r]));
            mt = rowmax16(mt);
            float mn = fmaxf(m[r], mt);
            float alpha = __expf(m[r] - mn);
            float p0 = __expf(s[0][r] - mn);
            float p1 = __expf(s[1][r] - mn);
            float p2 = __expf(s[2][r] - mn);
            float p3 = __expf(s[3][r] - mn);
            l[r] = l[r] * alpha + rowsum16(p0 + p1 + p2 + p3);
            o0[r] *= alpha;
            o1[r] *= alpha;
            m[r] = mn;
            const int row = hi * 8 + r;        // C layout -> row-major LDS
            pW[row * KBLK + mrow]      = (_Float16)p0;
            pW[row * KBLK + 16 + mrow] = (_Float16)p1;
            pW[row * KBLK + 32 + mrow] = (_Float16)p2;
            pW[row * KBLK + 48 + mrow] = (_Float16)p3;
        }

        // P@V over the 64-key block: two 32-key sub-blocks, 4 WMMAs
        #pragma unroll
        for (int c = 0; c < 2; c++) {
            v16h ap  = make_a(pW + mrow * KBLK + c * 32, base8);
            v16h bv0 = *(const v16h*)(vB + mrow * KBLK + c * 32 + hi * 16);
            v16h bv1 = *(const v16h*)(vB + (16 + mrow) * KBLK + c * 32 + hi * 16);
            o0 = __builtin_amdgcn_wmma_f32_16x16x32_f16(false, ap, false, bv0, (short)0, o0, false, false);
            o1 = __builtin_amdgcn_wmma_f32_16x16x32_f16(false, ap, false, bv1, (short)0, o1, false, false);
        }
        __syncthreads();                       // all waves done with kB/vB
    }

    // normalize and write to concat layout [B,S,D]
    const size_t obase = ((size_t)(b * SEQ + qrow)) * DMODEL + h * HD;
    #pragma unroll
    for (int r = 0; r < 8; r++) {
        const int row = hi * 8 + r;
        const float inv = 1.0f / l[r];
        attn[obase + (size_t)row * DMODEL + mrow]      = (_Float16)(o0[r] * inv);
        attn[obase + (size_t)row * DMODEL + 16 + mrow] = (_Float16)(o1[r] * inv);
    }
}

// ---------- output projection: 32x32 register tile per wave ----------
__global__ __launch_bounds__(32)
void proj_kernel(const _Float16* __restrict__ attn, const _Float16* __restrict__ woT,
                 const float* __restrict__ bo, float* __restrict__ out) {
    const int lane = threadIdx.x;
    const int mrow = lane & 15, hi = lane >> 4, base8 = hi * 8;
    const int idx = blockIdx.x;
    const int nt = idx & 31;    // 1024/32 col tiles
    const int mt = idx >> 5;    // 4096/32 row tiles
    const _Float16* arow0 = attn + (size_t)(mt * 32 + mrow) * DMODEL;
    const _Float16* arow1 = arow0 + (size_t)16 * DMODEL;
    const _Float16* wrow0 = woT + (size_t)(nt * 32 + mrow) * DMODEL;
    const _Float16* wrow1 = wrow0 + (size_t)16 * DMODEL;

    v8f c00 = {}, c01 = {}, c10 = {}, c11 = {};
    for (int kb = 0; kb < DMODEL; kb += 32) {
        v16h a0 = make_a(arow0 + kb, base8);
        v16h a1 = make_a(arow1 + kb, base8);
        v16h b0 = *(const v16h*)(wrow0 + kb + hi * 16);
        v16h b1 = *(const v16h*)(wrow1 + kb + hi * 16);
        c00 = __builtin_amdgcn_wmma_f32_16x16x32_f16(false, a0, false, b0, (short)0, c00, false, false);
        c01 = __builtin_amdgcn_wmma_f32_16x16x32_f16(false, a0, false, b1, (short)0, c01, false, false);
        c10 = __builtin_amdgcn_wmma_f32_16x16x32_f16(false, a1, false, b0, (short)0, c10, false, false);
        c11 = __builtin_amdgcn_wmma_f32_16x16x32_f16(false, a1, false, b1, (short)0, c11, false, false);
    }
    const float bias0 = bo[nt * 32 + mrow];
    const float bias1 = bo[nt * 32 + 16 + mrow];
    #pragma unroll
    for (int r = 0; r < 8; r++) {
        const int row0 = mt * 32 + hi * 8 + r;
        const int row1 = row0 + 16;
        float* o0p = out + (size_t)row0 * DMODEL + nt * 32;
        float* o1p = out + (size_t)row1 * DMODEL + nt * 32;
        o0p[mrow]      = c00[r] + bias0;
        o0p[16 + mrow] = c01[r] + bias1;
        o1p[mrow]      = c10[r] + bias0;
        o1p[16 + mrow] = c11[r] + bias1;
    }
}

// ---------- launch ----------
extern "C" void kernel_launch(void* const* d_in, const int* in_sizes, int n_in,
                              void* d_out, int out_size, void* d_ws, size_t ws_size,
                              hipStream_t stream) {
    const float* x  = (const float*)d_in[0];
    const float* wq = (const float*)d_in[1];
    const float* bq = (const float*)d_in[2];
    const float* wk = (const float*)d_in[3];
    const float* bk = (const float*)d_in[4];
    const float* wv = (const float*)d_in[5];
    const float* bv = (const float*)d_in[6];
    const float* wo = (const float*)d_in[7];
    const float* bo = (const float*)d_in[8];
    float* out = (float*)d_out;

    // workspace carve-up (halves); total ~35.7 MB
    _Float16* w   = (_Float16*)d_ws;
    _Float16* Q   = w;                                   // 4M halves
    _Float16* Kb  = w + (size_t)4 * 1024 * 1024;         // 4M
    _Float16* Vt  = w + (size_t)8 * 1024 * 1024;         // 4M
    _Float16* att = w + (size_t)12 * 1024 * 1024;        // 4M
    _Float16* wqT = w + (size_t)16 * 1024 * 1024;        // 1K
    _Float16* wkT = wqT + 1024;
    _Float16* wvT = wkT + 1024;
    _Float16* woT = wvT + 1024;                          // 1M halves

    prep_small<<<1, 1024, 0, stream>>>(wq, wk, wv, wqT, wkT, wvT);
    prep_wo<<<4096, 256, 0, stream>>>(wo, woT);
    qkv_kernel<<<BATCH * NH * QTILES, 32, 0, stream>>>(x, wqT, bq, wkT, bk, wvT, bv,
                                                       Q, Kb, Vt);
    attn_kernel<<<BATCH * NH * (SEQ / 128), 256, 0, stream>>>(Q, Kb, Vt, att);
    proj_kernel<<<(BATCH * SEQ / 32) * (DMODEL / 32), 32, 0, stream>>>(att, woT, bo, out);
}
